// patch_loss_64725157150820
// MI455X (gfx1250) — compile-verified, matrix operand order checked by reading, not run on hardware
//
#include <hip/hip_runtime.h>
#include <hip/hip_bf16.h>

typedef __attribute__((ext_vector_type(8))) int v8i;

#define NPC 524288           // 64 * 64 * 128 elements of pc
#define LOSS_BLOCKS 256
#define COUNT_BLOCKS 512     // 512 blocks * 8 waves = 4096 rows of 1024 int32

// ---------------------------------------------------------------------------
// Kernel 0: zero the 64 per-cell counters each call (ws is not re-poisoned).
// ---------------------------------------------------------------------------
__global__ void pl_zero_kernel(int* __restrict__ counts) {
    counts[threadIdx.x] = 0;   // launched with 64 threads
}

// ---------------------------------------------------------------------------
// Kernel 1: count (targets == 2) per 8x8 cell using V_WMMA_I32_16X16X64_IU8.
// One wave processes one (b, r) row of 1024 int32 (b in [0,64), r in [0,64)).
// Each lane packs 32 match-bytes (0/1) into a v8i A fragment (1024 bytes per
// wave). With B = all-ones bytes, sum over ALL entries of A x B equals
// 16 * (number of set match bytes), independent of operand striping.
// ---------------------------------------------------------------------------
__global__ void pl_count_kernel(const int* __restrict__ targets,
                                int* __restrict__ counts) {
    const int lane  = threadIdx.x & 31;
    const int wid   = threadIdx.x >> 5;               // 0..7
    const int gwave = blockIdx.x * 8 + wid;           // 0..4095
    const int b     = gwave >> 6;                     // batch index  0..63
    const int r     = gwave & 63;                     // H row index  0..63
    const int cell  = ((b >> 3) << 3) | (r >> 3);     // i*8 + j

    const int4* row = (const int4*)(targets + ((size_t)b * 512 + r) * 1024);

    v8i a;
    #pragma unroll
    for (int i = 0; i < 8; ++i) {
        // each load instruction: 32 lanes x 16B = 512 contiguous bytes
        const int4 v = row[i * 32 + lane];
        unsigned m = (unsigned)(v.x == 2)
                   | ((unsigned)(v.y == 2) << 8)
                   | ((unsigned)(v.z == 2) << 16)
                   | ((unsigned)(v.w == 2) << 24);
        a[i] = (int)m;
    }

    v8i ones;
    #pragma unroll
    for (int i = 0; i < 8; ++i) ones[i] = 0x01010101;

    v8i c = {};
    // 7 args: (sgn_a, A, sgn_b, B, C, reuse_a, reuse_b) -> v_wmma_i32_16x16x64_iu8
    c = __builtin_amdgcn_wmma_i32_16x16x64_iu8(false, a, false, ones, c,
                                               false, false);

    int s = 0;
    #pragma unroll
    for (int i = 0; i < 8; ++i) s += c[i];

    // deterministic wave32 butterfly: s becomes sum over all 32 lanes = 16*count
    #pragma unroll
    for (int off = 16; off > 0; off >>= 1) s += __shfl_xor(s, off, 32);

    if (lane == 0) atomicAdd(&counts[cell], s >> 4);  // exact: divide by 16
}

// ---------------------------------------------------------------------------
// Kernel 2: BCE partial sums. pc is (B=64, 1, h=64, w=128) flat.
// Each thread handles 8 consecutive floats -> shares b and hh, ww = ww0..ww0+7.
// t can be nonzero only when hh < 8 and ww0 == 0.
// ---------------------------------------------------------------------------
__global__ void pl_loss_kernel(const float* __restrict__ pc,
                               const int* __restrict__ counts,
                               float* __restrict__ partials) {
    __shared__ float condf[64];
    __shared__ float red[8];

    if (threadIdx.x < 64) {
        const int c = counts[threadIdx.x];
        condf[threadIdx.x] = (c > 0 && c < 64) ? 1.0f : 0.0f;
    }
    __syncthreads();

    const int base = (blockIdx.x * 256 + threadIdx.x) * 8;  // < 524288
    const int hh   = (base & 8191) >> 7;                    // row in h
    const bool special = (hh < 8) && ((base & 127) == 0);   // ww0 == 0

    const float4* p4 = (const float4*)(pc + base);
    float p[8];
    { float4 v = p4[0]; p[0]=v.x; p[1]=v.y; p[2]=v.z; p[3]=v.w; }
    { float4 v = p4[1]; p[4]=v.x; p[5]=v.y; p[6]=v.z; p[7]=v.w; }

    float acc = 0.0f;
    #pragma unroll
    for (int k = 0; k < 8; ++k) {
        const bool t1 = special && (condf[hh * 8 + k] != 0.0f); // t==1
        acc += t1 ? __logf(p[k]) * 0.0f + logf(p[k]) : log1pf(-p[k]);
    }

    // deterministic in-wave butterfly, then fixed-order cross-wave sum
    #pragma unroll
    for (int off = 16; off > 0; off >>= 1) acc += __shfl_xor(acc, off, 32);
    if ((threadIdx.x & 31) == 0) red[threadIdx.x >> 5] = acc;
    __syncthreads();
    if (threadIdx.x == 0) {
        float s = 0.0f;
        #pragma unroll
        for (int i = 0; i < 8; ++i) s += red[i];
        partials[blockIdx.x] = s;   // plain store, fully overwritten each call
    }
}

// ---------------------------------------------------------------------------
// Kernel 3: deterministic tree over 256 partials, final scale + negate.
// ---------------------------------------------------------------------------
__global__ void pl_final_kernel(const float* __restrict__ partials,
                                float* __restrict__ out) {
    __shared__ float s[256];
    s[threadIdx.x] = partials[threadIdx.x];
    __syncthreads();
    #pragma unroll
    for (int off = 128; off > 0; off >>= 1) {
        if (threadIdx.x < off) s[threadIdx.x] += s[threadIdx.x + off];
        __syncthreads();
    }
    if (threadIdx.x == 0) out[0] = -s[0] * (1.0f / (float)NPC);
}

// ---------------------------------------------------------------------------
extern "C" void kernel_launch(void* const* d_in, const int* in_sizes, int n_in,
                              void* d_out, int out_size, void* d_ws, size_t ws_size,
                              hipStream_t stream) {
    const float* pc      = (const float*)d_in[0];  // (64,1,64,128) f32
    const int*   targets = (const int*)d_in[1];    // (64,512,1024) i32

    int*   counts   = (int*)d_ws;                         // 64 ints
    float* partials = (float*)((char*)d_ws + 256);        // 256 floats

    pl_zero_kernel <<<1, 64, 0, stream>>>(counts);
    pl_count_kernel<<<COUNT_BLOCKS, 256, 0, stream>>>(targets, counts);
    pl_loss_kernel <<<LOSS_BLOCKS, 256, 0, stream>>>(pc, counts, partials);
    pl_final_kernel<<<1, 256, 0, stream>>>(partials, (float*)d_out);
}
